// XLNetSpanLabeling_69063074120362
// MI455X (gfx1250) — matrix operand order repaired and checked by reading, not compiled, not run on hardware
//
#include <hip/hip_runtime.h>
#include <hip/hip_bf16.h>

// ---------------------------------------------------------------------------
// XLNet span labeling head for MI455X (gfx1250, wave32, WMMA).
// B=16, L=512, H=1024, K=5.
// Strategy: pre-split fp32 -> bf16 hi/lo planes once (memory bound), run the
// dominant GEMM purely on v_wmma_f32_16x16x32_bf16 (3 WMMAs emulate fp32),
// fragments fetched as ds_load_b128 pairs from layout-matched LDS tiles.
// Workspace requirement ~76MB (see layout in kernel_launch).
// ---------------------------------------------------------------------------

typedef __attribute__((ext_vector_type(16))) __bf16 v16bf;
typedef __attribute__((ext_vector_type(8)))  __bf16 v8bf;
typedef __attribute__((ext_vector_type(8)))  float  v8f;

#define HDIM 1024
#define LSEQ 512
#define NBATCH 16
#define KTOP 5
#define NEGC 1e30f
#define LDST 40   // LDS row stride in bf16 elements (80B = 5*16B, keeps b128 alignment)

__device__ inline v16bf cat16(v8bf a, v8bf b) {
  return __builtin_shufflevector(a, b, 0,1,2,3,4,5,6,7,8,9,10,11,12,13,14,15);
}

// ---------------- reductions ----------------
__device__ inline float wave_sum(float v) {
#pragma unroll
  for (int o = 16; o; o >>= 1) v += __shfl_xor(v, o, 32);
  return v;
}

__device__ inline float br_max512(float v, float* sh, int tid) {
  sh[tid] = v; __syncthreads();
#pragma unroll
  for (int s = 256; s > 0; s >>= 1) {
    if (tid < s) sh[tid] = fmaxf(sh[tid], sh[tid + s]);
    __syncthreads();
  }
  float r = sh[0]; __syncthreads();
  return r;
}

__device__ inline float br_sum512(float v, float* sh, int tid) {
  sh[tid] = v; __syncthreads();
#pragma unroll
  for (int s = 256; s > 0; s >>= 1) {
    if (tid < s) sh[tid] += sh[tid + s];
    __syncthreads();
  }
  float r = sh[0]; __syncthreads();
  return r;
}

// ---------------------------------------------------------------------------
// Kernel 0: stream-split fp32 tensor into bf16 hi + lo residual planes.
// Pure memory-bound pass; hi = bf16(x), lo = bf16(x - hi).
// ---------------------------------------------------------------------------
__global__ __launch_bounds__(256) void k_split_bf16(
    const float* __restrict__ X, __bf16* __restrict__ Hi,
    __bf16* __restrict__ Lo, int n)
{
  for (int i = blockIdx.x * 256 + threadIdx.x; i < n; i += gridDim.x * 256) {
    float x = X[i];
    __bf16 h = (__bf16)x;
    Hi[i] = h;
    Lo[i] = (__bf16)(x - (float)h);
  }
}

// ---------------------------------------------------------------------------
// Kernel 1: per-batch start-stage. One block (512 threads = 16 waves) per b.
// Emits SF as bf16 hi/lo planes for the downstream WMMA GEMM.
// ---------------------------------------------------------------------------
__global__ __launch_bounds__(512) void k_start_stage(
    const float* __restrict__ seq, const int* __restrict__ cidx,
    const float* __restrict__ mask, const float* __restrict__ Wst,
    const float* __restrict__ bst, const float* __restrict__ Wansi,
    const float* __restrict__ bansi, const float* __restrict__ Wanso,
    float* __restrict__ out_sp, float* __restrict__ out_msl,
    float* __restrict__ out_tops, float* __restrict__ out_class,
    __bf16* __restrict__ SFhi, __bf16* __restrict__ SFlo)
{
  __shared__ float sl[LSEQ], work[LSEQ], prob[LSEQ];
  __shared__ float wst[HDIM], sfeat[HDIM], cfeat[HDIM];
  __shared__ float shv[LSEQ];
  __shared__ int   shi[LSEQ];
  __shared__ int   topidx[KTOP];

  const int b = blockIdx.x, tid = threadIdx.x;
  const int lane = tid & 31, wave = tid >> 5;

  for (int h = tid; h < HDIM; h += 512) wst[h] = Wst[h];
  __syncthreads();

  // 1) start logits: one wave per row, 32 rows per wave
  const float b0 = bst[0];
  for (int l = wave * 32; l < wave * 32 + 32; ++l) {
    const float* row = seq + ((size_t)b * LSEQ + l) * HDIM;
    float s = 0.f;
    for (int h = lane; h < HDIM; h += 32) s = fmaf(row[h], wst[h], s);
    s = wave_sum(s);
    if (lane == 0) {
      float m   = mask[b * LSEQ + l];
      float msl = (s + b0) * m - NEGC * (1.f - m);
      sl[l] = msl;
      out_msl[b * LSEQ + l] = msl;
    }
  }
  __syncthreads();

  // 2) log_softmax + softmax over L
  float msl = sl[tid];
  float mx  = br_max512(msl, shv, tid);
  float ex  = expf(msl - mx);
  float s   = br_sum512(ex, shv, tid);
  float lse = mx + logf(s);
  out_sp[b * LSEQ + tid] = msl - lse;
  prob[tid] = ex / s;
  work[tid] = msl;
  __syncthreads();

  // 3) top-5 by iterative argmax (tie -> lowest index)
  for (int j = 0; j < KTOP; ++j) {
    shv[tid] = work[tid]; shi[tid] = tid; __syncthreads();
#pragma unroll
    for (int st = 256; st > 0; st >>= 1) {
      if (tid < st) {
        float ov = shv[tid + st]; int oi = shi[tid + st];
        if (ov > shv[tid] || (ov == shv[tid] && oi < shi[tid])) { shv[tid] = ov; shi[tid] = oi; }
      }
      __syncthreads();
    }
    if (tid == 0) {
      topidx[j] = shi[0];
      out_tops[b * KTOP + j] = shv[0] - lse;
      work[shi[0]] = -3.0e38f;
    }
    __syncthreads();
  }

  // 4) gather start_features directly as split bf16 planes
  for (int t = tid; t < KTOP * HDIM; t += 512) {
    int k = t >> 10, h = t & (HDIM - 1);
    float x = seq[((size_t)b * LSEQ + topidx[k]) * HDIM + h];
    __bf16 hi = (__bf16)x;
    size_t o = ((size_t)b * KTOP + k) * HDIM + h;
    SFhi[o] = hi;
    SFlo[o] = (__bf16)(x - (float)hi);
  }

  // 5) softmax-pooled start feature + class feature
  const int ci = cidx[b];
  for (int h = tid; h < HDIM; h += 512) {
    float acc = 0.f;
    const float* col = seq + (size_t)b * LSEQ * HDIM + h;
    for (int l = 0; l < LSEQ; ++l) acc = fmaf(col[(size_t)l * HDIM], prob[l], acc);
    sfeat[h] = acc;
    cfeat[h] = seq[((size_t)b * LSEQ + ci) * HDIM + h];
  }
  __syncthreads();

  // 6) answer MLP
  float cpart = 0.f;
  for (int d = tid; d < HDIM; d += 512) {
    float acc = bansi[d];
    for (int h = 0; h < HDIM; ++h) acc = fmaf(sfeat[h], Wansi[(size_t)h * HDIM + d], acc);
    for (int h = 0; h < HDIM; ++h) acc = fmaf(cfeat[h], Wansi[(size_t)(HDIM + h) * HDIM + d], acc);
    cpart = fmaf(tanhf(acc), Wanso[d], cpart);
  }
  float cl = br_sum512(cpart, shv, tid);
  if (tid == 0) out_class[b] = cl;
}

// ---------------------------------------------------------------------------
// Kernel 2: split-precision WMMA GEMM on pre-split bf16 planes.
//   C[M,1024] = A[M,1024] @ B[1024,1024],  A ~ Ahi+Alo, B ~ Bhi+Blo.
// 256 threads = 8 waves (2 in M x 4 in N), wave tile 16x32 -> block 32x128.
// LDS: A planes row-major [32][LDST], B planes TRANSPOSED [128][LDST] so every
// fragment is two contiguous 16B chunks -> ds_load_b128 pairs, zero VALU.
// 6 v_wmma_f32_16x16x32_bf16 per wave per K-step (hi*hi + hi*lo + lo*hi).
// ---------------------------------------------------------------------------
__global__ __launch_bounds__(256) void k_gemm_bf16split(
    const __bf16* __restrict__ Ahi, const __bf16* __restrict__ Alo,
    const __bf16* __restrict__ Bhi, const __bf16* __restrict__ Blo,
    float* __restrict__ C, int M)
{
  const int N = 1024, K = 1024;
  __shared__ __bf16 sAh[32 * LDST], sAl[32 * LDST];
  __shared__ __bf16 sBh[128 * LDST], sBl[128 * LDST];

  const int tid  = threadIdx.x;
  const int lane = tid & 31, wave = tid >> 5;
  const int mi = wave >> 2, ni = wave & 3;       // 2 x 4 wave grid
  const int mBase = blockIdx.y * 32;
  const int nBase = blockIdx.x * 128;
  const int half = lane >> 4, q = lane & 15;

  // fragment chunk pointers (constant across the K loop)
  const v8bf* pAh0 = (const v8bf*)&sAh[(mi * 16 + q) * LDST + half * 8];
  const v8bf* pAh1 = (const v8bf*)&sAh[(mi * 16 + q) * LDST + 16 + half * 8];
  const v8bf* pAl0 = (const v8bf*)&sAl[(mi * 16 + q) * LDST + half * 8];
  const v8bf* pAl1 = (const v8bf*)&sAl[(mi * 16 + q) * LDST + 16 + half * 8];
  const v8bf* pBh00 = (const v8bf*)&sBh[(ni * 32 + q) * LDST + half * 8];
  const v8bf* pBh01 = (const v8bf*)&sBh[(ni * 32 + q) * LDST + 16 + half * 8];
  const v8bf* pBl00 = (const v8bf*)&sBl[(ni * 32 + q) * LDST + half * 8];
  const v8bf* pBl01 = (const v8bf*)&sBl[(ni * 32 + q) * LDST + 16 + half * 8];
  const v8bf* pBh10 = (const v8bf*)&sBh[(ni * 32 + 16 + q) * LDST + half * 8];
  const v8bf* pBh11 = (const v8bf*)&sBh[(ni * 32 + 16 + q) * LDST + 16 + half * 8];
  const v8bf* pBl10 = (const v8bf*)&sBl[(ni * 32 + 16 + q) * LDST + half * 8];
  const v8bf* pBl11 = (const v8bf*)&sBl[(ni * 32 + 16 + q) * LDST + 16 + half * 8];

  v8f acc0 = {}, acc1 = {};

  for (int k0 = 0; k0 < K; k0 += 32) {
    // ---- stage A planes: 32x32 bf16 each, dword copies (no conversion) ----
#pragma unroll
    for (int i = 0; i < 2; ++i) {
      int e = tid + i * 256;                  // 512 dwords per plane
      int r = e >> 4, c2 = e & 15;
      int gr = mBase + r;
      uint32_t vh = 0, vl = 0;
      if (gr < M) {
        vh = *(const uint32_t*)(Ahi + (size_t)gr * K + k0 + c2 * 2);
        vl = *(const uint32_t*)(Alo + (size_t)gr * K + k0 + c2 * 2);
      }
      *(uint32_t*)&sAh[r * LDST + c2 * 2] = vh;
      *(uint32_t*)&sAl[r * LDST + c2 * 2] = vl;
    }
    // ---- stage B planes transposed: 32(k) x 128(n) -> [n][k] ----
#pragma unroll
    for (int i = 0; i < 8; ++i) {
      int e = tid + i * 256;                  // 2048 dwords per plane
      int r = e >> 6, c2 = e & 63;            // k row, n-pair
      union { uint32_t u; __bf16 h[2]; } ch, cl2;
      ch.u  = *(const uint32_t*)(Bhi + (size_t)(k0 + r) * N + nBase + c2 * 2);
      cl2.u = *(const uint32_t*)(Blo + (size_t)(k0 + r) * N + nBase + c2 * 2);
      int n0 = c2 * 2;
      sBh[n0 * LDST + r]       = ch.h[0];
      sBh[(n0 + 1) * LDST + r] = ch.h[1];
      sBl[n0 * LDST + r]       = cl2.h[0];
      sBl[(n0 + 1) * LDST + r] = cl2.h[1];
    }
    __syncthreads();

    // ---- fragments: pure ds_load_b128 pairs ----
    v16bf ah  = cat16(*pAh0, *pAh1);
    v16bf al  = cat16(*pAl0, *pAl1);
    v16bf bh0 = cat16(*pBh00, *pBh01);
    v16bf bl0 = cat16(*pBl00, *pBl01);
    v16bf bh1 = cat16(*pBh10, *pBh11);
    v16bf bl1 = cat16(*pBl10, *pBl11);

    acc0 = __builtin_amdgcn_wmma_f32_16x16x32_bf16(false, ah, false, bh0, (short)0, acc0, false, false);
    acc0 = __builtin_amdgcn_wmma_f32_16x16x32_bf16(false, ah, false, bl0, (short)0, acc0, false, false);
    acc0 = __builtin_amdgcn_wmma_f32_16x16x32_bf16(false, al, false, bh0, (short)0, acc0, false, false);
    acc1 = __builtin_amdgcn_wmma_f32_16x16x32_bf16(false, ah, false, bh1, (short)0, acc1, false, false);
    acc1 = __builtin_amdgcn_wmma_f32_16x16x32_bf16(false, ah, false, bl1, (short)0, acc1, false, false);
    acc1 = __builtin_amdgcn_wmma_f32_16x16x32_bf16(false, al, false, bh1, (short)0, acc1, false, false);
    __syncthreads();
  }

  // store: C/D layout VGPR r -> row r + 8*half, col = lane&15
#pragma unroll
  for (int r = 0; r < 8; ++r) {
    int row = mBase + mi * 16 + r + half * 8;
    if (row < M) {
      int col = nBase + ni * 32 + q;
      C[(size_t)row * N + col]      = acc0[r];
      C[(size_t)row * N + col + 16] = acc1[r];
    }
  }
}

// ---------------------------------------------------------------------------
// Kernel 3: fused tanh + LayerNorm + projection -> end logits; (B,L,K,H)
// tensor never materialized. One wave per (b,l); P1 row in registers reused
// over k; P2 rows + gamma*Wout + bias in LDS.
// ---------------------------------------------------------------------------
__global__ __launch_bounds__(256) void k_end_logits(
    const float* __restrict__ P1, const float* __restrict__ P2,
    const float* __restrict__ bendi, const float* __restrict__ gam,
    const float* __restrict__ bet, const float* __restrict__ Wout,
    const float* __restrict__ bendo, float* __restrict__ EL)
{
  __shared__ float P2s[KTOP * HDIM];
  __shared__ float gwS[HDIM], bS[HDIM];
  __shared__ float red[256];
  __shared__ float sgwS, sbwS;

  const int tid = threadIdx.x, lane = tid & 31, wave = tid >> 5;
  const int b  = blockIdx.x >> 6;
  const int l0 = (blockIdx.x & 63) << 3;

  for (int i = tid; i < KTOP * HDIM; i += 256)
    P2s[i] = P2[(size_t)b * KTOP * HDIM + i];

  float sgwp = 0.f, sbwp = 0.f;
  for (int h = tid; h < HDIM; h += 256) {
    float w = Wout[h];
    gwS[h] = gam[h] * w;
    bS[h]  = bendi[h];
    sgwp  = fmaf(gam[h], w, sgwp);
    sbwp  = fmaf(bet[h], w, sbwp);
  }
  red[tid] = sgwp; __syncthreads();
#pragma unroll
  for (int s = 128; s > 0; s >>= 1) { if (tid < s) red[tid] += red[tid + s]; __syncthreads(); }
  if (tid == 0) sgwS = red[0];
  __syncthreads();
  red[tid] = sbwp; __syncthreads();
#pragma unroll
  for (int s = 128; s > 0; s >>= 1) { if (tid < s) red[tid] += red[tid + s]; __syncthreads(); }
  if (tid == 0) sbwS = red[0];
  __syncthreads();

  const int l = l0 + wave;
  const float* p1row = P1 + ((size_t)b * LSEQ + l) * HDIM;
  float p1r[32];
#pragma unroll
  for (int j = 0; j < 32; ++j) p1r[j] = p1row[lane + 32 * j];

  const float be = bendo[0];
  for (int k = 0; k < KTOP; ++k) {
    float s1 = 0.f, s2 = 0.f, s3 = 0.f;
#pragma unroll
    for (int j = 0; j < 32; ++j) {
      int h = lane + 32 * j;
      float v = tanhf(p1r[j] + P2s[k * HDIM + h] + bS[h]);
      s1 += v;
      s2 = fmaf(v, v, s2);
      s3 = fmaf(v, gwS[h], s3);
    }
    s1 = wave_sum(s1); s2 = wave_sum(s2); s3 = wave_sum(s3);
    if (lane == 0) {
      float mu  = s1 * (1.f / HDIM);
      float var = s2 * (1.f / HDIM) - mu * mu;
      float rs  = rsqrtf(var + 1e-12f);
      EL[((size_t)b * KTOP + k) * LSEQ + l] = (s3 - mu * sgwS) * rs + sbwS + be;
    }
  }
}

// ---------------------------------------------------------------------------
// Kernel 4: per (b,k): mask, log_softmax over L, top-5.
// ---------------------------------------------------------------------------
__global__ __launch_bounds__(512) void k_end_final(
    const float* __restrict__ EL, const float* __restrict__ mask,
    float* __restrict__ out_ep, float* __restrict__ out_mel,
    float* __restrict__ out_tope)
{
  __shared__ float work[LSEQ], shv[LSEQ];
  __shared__ int   shi[LSEQ];
  const int bk = blockIdx.x, b = bk / KTOP, k = bk % KTOP, tid = threadIdx.x;

  float raw = EL[(size_t)bk * LSEQ + tid];
  float m   = mask[b * LSEQ + tid];
  float mel = raw * m - NEGC * (1.f - m);
  out_mel[(size_t)bk * LSEQ + tid] = mel;

  float mx  = br_max512(mel, shv, tid);
  float ex  = expf(mel - mx);
  float s   = br_sum512(ex, shv, tid);
  float lse = mx + logf(s);
  out_ep[(size_t)bk * LSEQ + tid] = mel - lse;
  work[tid] = mel; __syncthreads();

  for (int j = 0; j < KTOP; ++j) {
    shv[tid] = work[tid]; shi[tid] = tid; __syncthreads();
#pragma unroll
    for (int st = 256; st > 0; st >>= 1) {
      if (tid < st) {
        float ov = shv[tid + st]; int oi = shi[tid + st];
        if (ov > shv[tid] || (ov == shv[tid] && oi < shi[tid])) { shv[tid] = ov; shi[tid] = oi; }
      }
      __syncthreads();
    }
    if (tid == 0) {
      out_tope[b * (KTOP * KTOP) + k * KTOP + j] = shv[0] - lse;
      work[shi[0]] = -3.0e38f;
    }
    __syncthreads();
  }
}

// ---------------------------------------------------------------------------
extern "C" void kernel_launch(void* const* d_in, const int* in_sizes, int n_in,
                              void* d_out, int out_size, void* d_ws, size_t ws_size,
                              hipStream_t stream) {
  const float* seq   = (const float*)d_in[0];   // (B,L,H)
  const int*   cidx  = (const int*)  d_in[1];   // (B,)
  const float* mask  = (const float*)d_in[2];   // (B,L)
  const float* Wst   = (const float*)d_in[3];   // (H,1)
  const float* bst   = (const float*)d_in[4];   // (1,)
  const float* Wend  = (const float*)d_in[5];   // (2H,H)
  const float* bendi = (const float*)d_in[6];   // (H,)
  const float* gam   = (const float*)d_in[7];   // (H,)
  const float* bet   = (const float*)d_in[8];   // (H,)
  const float* Wout  = (const float*)d_in[9];   // (H,1)
  const float* bendo = (const float*)d_in[10];  // (1,)
  const float* Wansi = (const float*)d_in[11];  // (2H,H)
  const float* bansi = (const float*)d_in[12];  // (H,)
  const float* Wanso = (const float*)d_in[13];  // (H,1)

  // output layout (flat concat, 98800 floats)
  float* out        = (float*)d_out;
  float* out_sp     = out;                         // (B,L)    8192
  float* out_ep     = out + 8192;                  // (B,K,L) 40960
  float* out_msl    = out + 49152;                 // (B,L)    8192
  float* out_mel    = out + 57344;                 // (B,K,L) 40960
  float* out_tops   = out + 98304;                 // (B,5)      80
  float* out_tope   = out + 98384;                 // (B,25)    400
  float* out_class  = out + 98784;                 // (B,)       16

  // workspace layout (~76MB)
  const size_t nSeq = (size_t)NBATCH * LSEQ * HDIM;   // 8,388,608
  const size_t nW   = (size_t)2 * HDIM * HDIM;        // 2,097,152
  const size_t nSF  = (size_t)NBATCH * KTOP * HDIM;   //    81,920
  char* wsb = (char*)d_ws;
  float*  P1    = (float*)wsb;                  wsb += nSeq * sizeof(float);      // 32MB
  float*  P2    = (float*)wsb;                  wsb += nSF * sizeof(float);
  float*  EL    = (float*)wsb;                  wsb += (size_t)NBATCH * KTOP * LSEQ * sizeof(float);
  __bf16* seqHi = (__bf16*)wsb;                 wsb += nSeq * sizeof(__bf16);     // 16MB
  __bf16* seqLo = (__bf16*)wsb;                 wsb += nSeq * sizeof(__bf16);     // 16MB
  __bf16* WHi   = (__bf16*)wsb;                 wsb += nW * sizeof(__bf16);       // 4MB
  __bf16* WLo   = (__bf16*)wsb;                 wsb += nW * sizeof(__bf16);       // 4MB
  __bf16* SFhi  = (__bf16*)wsb;                 wsb += nSF * sizeof(__bf16);
  __bf16* SFlo  = (__bf16*)wsb;

  // 0) pre-split fp32 -> bf16 hi/lo planes (streaming, memory-bound)
  k_split_bf16<<<4096, 256, 0, stream>>>(seq,  seqHi, seqLo, (int)nSeq);
  k_split_bf16<<<2048, 256, 0, stream>>>(Wend, WHi,   WLo,   (int)nW);

  // 1) start stage (produces SF hi/lo planes)
  k_start_stage<<<NBATCH, 512, 0, stream>>>(
      seq, cidx, mask, Wst, bst, Wansi, bansi, Wanso,
      out_sp, out_msl, out_tops, out_class, SFhi, SFlo);

  // 2) P1 = seq @ W1 ; P2 = SF @ W2   (split-precision bf16 WMMA)
  dim3 g1(HDIM / 128, (NBATCH * LSEQ) / 32);
  k_gemm_bf16split<<<g1, 256, 0, stream>>>(seqHi, seqLo, WHi, WLo, P1, NBATCH * LSEQ);
  dim3 g2(HDIM / 128, 3);
  k_gemm_bf16split<<<g2, 256, 0, stream>>>(
      SFhi, SFlo, WHi + (size_t)HDIM * HDIM, WLo + (size_t)HDIM * HDIM, P2, NBATCH * KTOP);

  // 3) fused tanh+LN+proj -> raw end logits (B,K,L)
  k_end_logits<<<NBATCH * (LSEQ / 8), 256, 0, stream>>>(
      P1, P2, bendi, gam, bet, Wout, bendo, EL);

  // 4) mask + log_softmax + top5 over L per (b,k)
  k_end_final<<<NBATCH * KTOP, 512, 0, stream>>>(EL, mask, out_ep, out_mel, out_tope);
}
